// GPTLanguageModel_65257733095793
// MI455X (gfx1250) — matrix operand order repaired and checked by reading, not compiled, not run on hardware
//
#include <hip/hip_runtime.h>
#include <hip/hip_bf16.h>
#include <math.h>

// ---------------------------------------------------------------------------
// GPT forward for MI455X (gfx1250, wave32, WMMA + Tensor Data Mover).
//  * all matmuls: v_wmma_f32_16x16x32_f16 (f16 operands, f32 accum)
//  * weights converted once per launch fp32->f16, pre-transposed to [N][K]
//    so every GEMM tile is a contiguous row-major f16 2D tile
//  * tiles DMA'd into padded LDS via tensor_load_to_lds (TENSORcnt),
//    double-buffered against the WMMA loop
//  * attention fused flash-style (online softmax), K tiles via TDM
// ---------------------------------------------------------------------------

typedef __attribute__((ext_vector_type(16))) _Float16 v16h;
typedef __attribute__((ext_vector_type(8)))  _Float16 v8h;
typedef __attribute__((ext_vector_type(8)))  float    v8f;
typedef __attribute__((ext_vector_type(4)))  unsigned tdm_u32x4;
typedef __attribute__((ext_vector_type(4)))  int      tdm_i32x4;
typedef __attribute__((ext_vector_type(8)))  int      tdm_i32x8;

union F8 { v8f v; float f[8]; };

// Model constants
#define BB 2
#define TT 1024
#define DD 768
#define HH 12
#define LL 8
#define VV 32000
#define BT (BB * TT)
#define D4 (4 * DD)

#if __has_builtin(__builtin_amdgcn_tensor_load_to_lds) && \
    __has_builtin(__builtin_amdgcn_s_wait_tensorcnt)
#define USE_TDM 1
#else
#define USE_TDM 0
#endif

// ---------------------------------------------------------------------------
// TDM: DMA a 64-row x 64-half tile (rows contiguous, row stride = rowlen
// elements) from global f16 into LDS with +8-half row padding (stride 72).
// D# fields per CDNA5 ISA 8.3/8.4: data_size=1(2B), pad_interval=4 (32 DWORDs
// = one 128B row), pad_amount=3 (4 DWORDs = 8 halves), type=2, count=1.
// ---------------------------------------------------------------------------
#if USE_TDM
__device__ __forceinline__ void tdm_load_tile_f16(const _Float16* src,
                                                  unsigned lds_byte_off,
                                                  int rowlen) {
  unsigned long long ga = (unsigned long long)(size_t)src;
  tdm_u32x4 g0;
  g0[0] = 1u;                                   // count=1, user descriptor
  g0[1] = lds_byte_off;                         // lds_addr
  g0[2] = (unsigned)ga;                         // global_addr[31:0]
  g0[3] = (unsigned)((ga >> 32) & 0x01FFFFFFull) | (2u << 30);  // [56:32]|type=2
  tdm_i32x8 g1;
  g1[0] = (1 << 16) | (1 << 20) | (4 << 22) | (3 << 25);
  g1[1] = (rowlen & 0xFFFF) << 16;              // tensor_dim0 lo
  g1[2] = ((unsigned)rowlen >> 16) | (64 << 16);// tensor_dim0 hi | tensor_dim1=64
  g1[3] = (64 << 16);                           // tile_dim0 = 64 elements
  g1[4] = 64;                                   // tile_dim1 = 64 rows
  g1[5] = rowlen;                               // tensor_dim0_stride lo32
  g1[6] = 0;
  g1[7] = 0;
  tdm_i32x4 z4 = {0, 0, 0, 0};
#if defined(__clang_major__) && __clang_major__ >= 23
  tdm_i32x8 z8 = {0, 0, 0, 0, 0, 0, 0, 0};
  __builtin_amdgcn_tensor_load_to_lds(g0, g1, z4, z4, z8, 0);
#else
  __builtin_amdgcn_tensor_load_to_lds(g0, g1, z4, z4, 0);
#endif
}
#endif

// ---------------------------------------------------------------------------
// WMMA helpers
// ---------------------------------------------------------------------------
__device__ __forceinline__ v8f wmma_f16(v16h a, v16h b, v8f c) {
  return __builtin_amdgcn_wmma_f32_16x16x32_f16(false, a, false, b,
                                                (short)0, c, false, false);
}

// A-operand fragment from LDS [m][k]: lanes 0-15 = M rows; lane-half h holds
// K chunks h*8..h*8+7 and h*8+16..h*8+23.
__device__ __forceinline__ v16h lds_frag_a(const _Float16* base, int mrow, int ks) {
  const int lane = threadIdx.x & 31;
  const _Float16* p = base + (mrow + (lane & 15)) * 72 + ks + ((lane >> 4) << 3);
  v8h lo = *(const v8h*)p;
  v8h hi = *(const v8h*)(p + 16);
  v16h f;
#pragma unroll
  for (int i = 0; i < 8; ++i) { f[i] = lo[i]; f[i + 8] = hi[i]; }
  return f;
}

// B-operand fragment from LDS [n][k]: lane holds column n=lane%16; lane-half h
// holds K chunks h*8..h*8+7 and h*8+16..h*8+23.
__device__ __forceinline__ v16h lds_frag_b(const _Float16* base, int nrow, int ks) {
  const int lane = threadIdx.x & 31;
  const _Float16* p = base + (nrow + (lane & 15)) * 72 + ks + ((lane >> 4) << 3);
  v8h lo = *(const v8h*)p;
  v8h hi = *(const v8h*)(p + 16);
  v16h f;
#pragma unroll
  for (int i = 0; i < 8; ++i) { f[i] = lo[i]; f[i + 8] = hi[i]; }
  return f;
}

// 16x64 strip of MMAs: acc[t] += A(16x64) * B(64x16 per tile t)
__device__ __forceinline__ void mma_tile(const _Float16* As, const _Float16* Bs,
                                         F8 acc[4], int wv) {
#pragma unroll
  for (int ks = 0; ks < 64; ks += 32) {
    v16h aF = lds_frag_a(As, wv * 16, ks);
#pragma unroll
    for (int t = 0; t < 4; ++t) {
      v16h bF = lds_frag_b(Bs, t * 16, ks);
      acc[t].v = wmma_f16(aF, bF, acc[t].v);
    }
  }
}

// ---------------------------------------------------------------------------
// Embedding: x[b,t,:] = tok_emb[idx[b,t],:] + pos_emb[t,:]  (fp32 residual)
// ---------------------------------------------------------------------------
__global__ void __launch_bounds__(256) embed_kernel(const int* __restrict__ idx,
                                                    const float* __restrict__ tok,
                                                    const float* __restrict__ pos,
                                                    float* __restrict__ x) {
  int i = blockIdx.x * 256 + threadIdx.x;
  if (i >= BT * DD) return;
  int d  = i % DD;
  int bt = i / DD;
  int t  = bt % TT;
  x[i] = tok[(size_t)idx[bt] * DD + d] + pos[t * DD + d];
}

// ---------------------------------------------------------------------------
// Weight convert+transpose: W fp32 [K][N] -> Wt f16 [N][K] (once per launch)
// ---------------------------------------------------------------------------
__global__ void __launch_bounds__(256) cvtT_kernel(const float* __restrict__ W,
                                                   _Float16* __restrict__ Wt,
                                                   int K, int N) {
  int i = blockIdx.x * 256 + threadIdx.x;
  if (i >= K * N) return;
  int k = i / N, n = i % N;
  Wt[(size_t)n * K + k] = (_Float16)W[i];
}

// ---------------------------------------------------------------------------
// LayerNorm over D=768, one row per block, f16 output (GEMM operand)
// ---------------------------------------------------------------------------
__global__ void __launch_bounds__(256) ln_kernel(const float* __restrict__ x,
                                                 const float* __restrict__ w,
                                                 const float* __restrict__ b,
                                                 _Float16* __restrict__ y) {
  const int row = blockIdx.x;
  const int tid = threadIdx.x;
  const float* xr = x + (size_t)row * DD;
  float s = 0.f, s2 = 0.f;
  for (int c = tid; c < DD; c += 256) { float v = xr[c]; s += v; s2 += v * v; }
  __shared__ float r1[256];
  __shared__ float r2[256];
  r1[tid] = s; r2[tid] = s2;
  __syncthreads();
  for (int off = 128; off > 0; off >>= 1) {
    if (tid < off) { r1[tid] += r1[tid + off]; r2[tid] += r2[tid + off]; }
    __syncthreads();
  }
  const float mean = r1[0] * (1.0f / DD);
  const float var  = r2[0] * (1.0f / DD) - mean * mean;
  const float inv  = rsqrtf(var + 1e-5f);
  for (int c = tid; c < DD; c += 256)
    y[(size_t)row * DD + c] = (_Float16)((xr[c] - mean) * inv * w[c] + b[c]);
}

// ---------------------------------------------------------------------------
// GEMM: C[M,N] = A[M,K](f16) * Bt[N,K](f16, pre-transposed weights)
// 128 threads / 4 waves, 64x64 tile, BK=64, TDM double-buffered LDS feed.
// Epilogue: optional bias / exact GELU / fp32 residual; OUT = f16 or fp32.
// ---------------------------------------------------------------------------
template <bool BIAS, bool RES, bool GELU, typename OUT>
__global__ void __launch_bounds__(128) gemm_kernel(const _Float16* __restrict__ A,
                                                   const _Float16* __restrict__ Bt,
                                                   const float* __restrict__ bias,
                                                   const float* __restrict__ Res,
                                                   OUT* __restrict__ C,
                                                   int M, int N, int K) {
  __shared__ __align__(16) _Float16 As[2][64][72];
  __shared__ __align__(16) _Float16 Bs[2][64][72];
  const int tid  = threadIdx.x;
  const int lane = tid & 31;
  const int wv   = tid >> 5;
  const int half = lane >> 4;
  const int lm   = lane & 15;
  const int m0 = blockIdx.y * 64;
  const int n0 = blockIdx.x * 64;

  F8 acc[4];
#pragma unroll
  for (int t = 0; t < 4; ++t)
#pragma unroll
    for (int i = 0; i < 8; ++i) acc[t].f[i] = 0.f;

#if USE_TDM
  if (tid == 0) {
    tdm_load_tile_f16(A  + (size_t)m0 * K, (unsigned)(size_t)&As[0][0][0], K);
    tdm_load_tile_f16(Bt + (size_t)n0 * K, (unsigned)(size_t)&Bs[0][0][0], K);
    __builtin_amdgcn_s_wait_tensorcnt(0);
  }
  __syncthreads();
  int nbuf = 0;
  for (int k0 = 0; k0 < K; k0 += 64) {
    const int cur = nbuf;
    nbuf ^= 1;
    if (tid == 0 && k0 + 64 < K) {  // prefetch next tiles via DMA
      tdm_load_tile_f16(A  + (size_t)m0 * K + k0 + 64,
                        (unsigned)(size_t)&As[nbuf][0][0], K);
      tdm_load_tile_f16(Bt + (size_t)n0 * K + k0 + 64,
                        (unsigned)(size_t)&Bs[nbuf][0][0], K);
    }
    mma_tile(&As[cur][0][0], &Bs[cur][0][0], acc, wv);
    if (tid == 0 && k0 + 64 < K) __builtin_amdgcn_s_wait_tensorcnt(0);
    __syncthreads();
  }
#else
  for (int k0 = 0; k0 < K; k0 += 64) {
#pragma unroll
    for (int e = 0; e < 4; ++e) {
      int chunk = e * 128 + tid;
      int r = chunk >> 3, c8 = (chunk & 7) * 8;
      *(v8h*)&As[0][r][c8] = *(const v8h*)&A[(size_t)(m0 + r) * K + k0 + c8];
      *(v8h*)&Bs[0][r][c8] = *(const v8h*)&Bt[(size_t)(n0 + r) * K + k0 + c8];
    }
    if (k0 + 64 < K) {
      __builtin_prefetch(&A[(size_t)(m0 + (tid & 63)) * K + k0 + 64], 0, 1);
      __builtin_prefetch(&Bt[(size_t)(n0 + (tid & 63)) * K + k0 + 64], 0, 1);
    }
    __syncthreads();
    mma_tile(&As[0][0][0], &Bs[0][0][0], acc, wv);
    __syncthreads();
  }
#endif

#pragma unroll
  for (int t = 0; t < 4; ++t)
#pragma unroll
    for (int i = 0; i < 8; ++i) {
      int row = m0 + wv * 16 + 8 * half + i;
      int col = n0 + t * 16 + lm;
      float val = acc[t].f[i];
      if (BIAS) val += bias[col];
      if (GELU) val = 0.5f * val * (1.0f + erff(val * 0.70710678118654752f));
      if (RES)  val += Res[(size_t)row * N + col];
      C[(size_t)row * N + col] = (OUT)val;
    }
}

// ---------------------------------------------------------------------------
// Fused causal attention (flash-style), f16 q/k/v/o, scale folded into S.
// One block = one (b,h) x 64 query rows; 4 waves x 16-row strips.
// ---------------------------------------------------------------------------
__global__ void __launch_bounds__(128) attn_kernel(const _Float16* __restrict__ q,
                                                   const _Float16* __restrict__ k,
                                                   const _Float16* __restrict__ v,
                                                   _Float16* __restrict__ o) {
  __shared__ __align__(16) _Float16 Qs[64][72];  // [qrow][hd]
  __shared__ __align__(16) _Float16 Ks[64][72];  // [key][hd]
  __shared__ __align__(16) _Float16 Vs[64][72];  // [hd][key] (transposed)
  __shared__ __align__(16) _Float16 Ps[64][72];  // [qrow][key]

  const int tid  = threadIdx.x;
  const int lane = tid & 31;
  const int wv   = tid >> 5;
  const int half = lane >> 4;
  const int lm   = lane & 15;
  const int bh = blockIdx.y;
  const int b  = bh / HH;
  const int hd = bh % HH;
  const int row0 = blockIdx.x * 64;
  const size_t base = (size_t)b * TT * DD + (size_t)hd * 64;
  const float scale = 0.03608439182435161f;  // D^-0.5 (reference scaling)

#pragma unroll
  for (int e = 0; e < 4; ++e) {  // Q tile, vectorized copy
    int chunk = e * 128 + tid;
    int r = chunk >> 3, c8 = (chunk & 7) * 8;
    *(v8h*)&Qs[r][c8] = *(const v8h*)&q[base + (size_t)(row0 + r) * DD + c8];
  }

  F8 oacc[4];
  float m_i[8], l_i[8];
#pragma unroll
  for (int t = 0; t < 4; ++t)
#pragma unroll
    for (int i = 0; i < 8; ++i) oacc[t].f[i] = 0.f;
#pragma unroll
  for (int i = 0; i < 8; ++i) { m_i[i] = -3.0e38f; l_i[i] = 0.f; }

  for (int jb = 0; jb <= blockIdx.x; ++jb) {
#if USE_TDM
    if (tid == 0)
      tdm_load_tile_f16(k + base + (size_t)(jb * 64) * DD,
                        (unsigned)(size_t)&Ks[0][0], DD);
#else
#pragma unroll
    for (int e = 0; e < 4; ++e) {
      int chunk = e * 128 + tid;
      int r = chunk >> 3, c8 = (chunk & 7) * 8;
      *(v8h*)&Ks[r][c8] = *(const v8h*)&k[base + (size_t)(jb * 64 + r) * DD + c8];
    }
#endif
    // V tile transposed (manual staging overlaps the K DMA)
#pragma unroll
    for (int e = 0; e < 32; ++e) {
      int flat = e * 128 + tid;
      int r = flat >> 6, c = flat & 63;
      Vs[c][r] = v[base + (size_t)(jb * 64 + r) * DD + c];
    }
    if (jb < blockIdx.x)
      __builtin_prefetch(&v[base + (size_t)((jb + 1) * 64 + (tid & 63)) * DD], 0, 1);
#if USE_TDM
    if (tid == 0) __builtin_amdgcn_s_wait_tensorcnt(0);
#endif
    __syncthreads();

    // S = Q * K^T
    F8 s[4];
#pragma unroll
    for (int t = 0; t < 4; ++t)
#pragma unroll
      for (int i = 0; i < 8; ++i) s[t].f[i] = 0.f;
    mma_tile(&Qs[0][0], &Ks[0][0], s, wv);

    // scale + causal mask
    const int colb = jb * 64 + lm;
    const int rowb = row0 + wv * 16 + 8 * half;
#pragma unroll
    for (int t = 0; t < 4; ++t)
#pragma unroll
      for (int i = 0; i < 8; ++i) {
        s[t].f[i] *= scale;
        if (colb + t * 16 > rowb + i) s[t].f[i] = -3.0e38f;
      }

    // online softmax (row lives across 16 lanes of a half-wave)
#pragma unroll
    for (int i = 0; i < 8; ++i) {
      float mx = fmaxf(fmaxf(s[0].f[i], s[1].f[i]), fmaxf(s[2].f[i], s[3].f[i]));
#pragma unroll
      for (int d = 1; d < 16; d <<= 1) mx = fmaxf(mx, __shfl_xor(mx, d, 32));
      float mnew  = fmaxf(m_i[i], mx);
      float alpha = __expf(m_i[i] - mnew);
      l_i[i] *= alpha;
#pragma unroll
      for (int t = 0; t < 4; ++t) oacc[t].f[i] *= alpha;
      float ps = 0.f;
#pragma unroll
      for (int t = 0; t < 4; ++t) {
        float p = __expf(s[t].f[i] - mnew);
        s[t].f[i] = p;
        ps += p;
      }
#pragma unroll
      for (int d = 1; d < 16; d <<= 1) ps += __shfl_xor(ps, d, 32);
      l_i[i] += ps;
      m_i[i] = mnew;
    }

    // C-layout -> A-layout via LDS
#pragma unroll
    for (int t = 0; t < 4; ++t)
#pragma unroll
      for (int i = 0; i < 8; ++i)
        Ps[wv * 16 + 8 * half + i][t * 16 + lm] = (_Float16)s[t].f[i];
    __syncthreads();

    // O += P * V
    mma_tile(&Ps[0][0], &Vs[0][0], oacc, wv);
    __syncthreads();
  }

#pragma unroll
  for (int t = 0; t < 4; ++t)
#pragma unroll
    for (int i = 0; i < 8; ++i) {
      int grow = row0 + wv * 16 + 8 * half + i;
      o[base + (size_t)grow * DD + t * 16 + lm] = (_Float16)(oacc[t].f[i] / l_i[i]);
    }
}

// ---------------------------------------------------------------------------
// Loss: per-row log-softmax NLL, mean over B*T rows.
// ---------------------------------------------------------------------------
__global__ void __launch_bounds__(1) zero_kernel(float* p) { *p = 0.f; }

__global__ void __launch_bounds__(256) loss_kernel(const float* __restrict__ logits,
                                                   const int* __restrict__ targets,
                                                   float* __restrict__ loss) {
  const int row = blockIdx.x;
  const int tid = threadIdx.x;
  const float* lr = logits + (size_t)row * VV;
  __shared__ float red[256];

  float mx = -3.0e38f;
  for (int c = tid; c < VV; c += 256) mx = fmaxf(mx, lr[c]);
  red[tid] = mx;
  __syncthreads();
  for (int off = 128; off > 0; off >>= 1) {
    if (tid < off) red[tid] = fmaxf(red[tid], red[tid + off]);
    __syncthreads();
  }
  mx = red[0];
  __syncthreads();

  float s = 0.f;
  for (int c = tid; c < VV; c += 256) s += __expf(lr[c] - mx);
  red[tid] = s;
  __syncthreads();
  for (int off = 128; off > 0; off >>= 1) {
    if (tid < off) red[tid] += red[tid + off];
    __syncthreads();
  }
  if (tid == 0) {
    float lse = mx + __logf(red[0]);
    float li  = (lse - lr[targets[row]]) * (1.0f / (float)BT);
    atomicAdd(loss, li);
  }
}

// ---------------------------------------------------------------------------
// Host launcher
// ---------------------------------------------------------------------------
extern "C" void kernel_launch(void* const* d_in, const int* in_sizes, int n_in,
                              void* d_out, int out_size, void* d_ws, size_t ws_size,
                              hipStream_t stream) {
  const int*   idx     = (const int*)d_in[0];
  const int*   targets = (const int*)d_in[1];
  const float* tok_emb = (const float*)d_in[2];
  const float* pos_emb = (const float*)d_in[3];
  const float* ln1_w   = (const float*)d_in[4];
  const float* ln1_b   = (const float*)d_in[5];
  const float* wq      = (const float*)d_in[6];
  const float* wk      = (const float*)d_in[7];
  const float* wvv     = (const float*)d_in[8];
  const float* wo      = (const float*)d_in[9];
  const float* bo      = (const float*)d_in[10];
  const float* ln2_w   = (const float*)d_in[11];
  const float* ln2_b   = (const float*)d_in[12];
  const float* w1      = (const float*)d_in[13];
  const float* b1      = (const float*)d_in[14];
  const float* w2      = (const float*)d_in[15];
  const float* b2      = (const float*)d_in[16];
  const float* lnf_w   = (const float*)d_in[17];
  const float* lnf_b   = (const float*)d_in[18];
  const float* lm_w    = (const float*)d_in[19];
  const float* lm_b    = (const float*)d_in[20];

  // workspace layout: fp32 residual stream, then f16 activations + weights
  float* x = (float*)d_ws;                                  // [BT, D] fp32
  _Float16* hb = (_Float16*)(x + (size_t)BT * DD);          // [BT, D]
  _Float16* qb = hb + (size_t)BT * DD;
  _Float16* kb = qb + (size_t)BT * DD;
  _Float16* vb = kb + (size_t)BT * DD;
  _Float16* ob = vb + (size_t)BT * DD;
  _Float16* mb = ob + (size_t)BT * DD;                      // [BT, 4D]
  _Float16* wt = mb + (size_t)BT * D4;                      // f16 weights
  const size_t PER = 4 * (size_t)DD * DD + 2 * (size_t)DD * D4;  // per layer
  _Float16* lm_wt = wt + (size_t)LL * PER;                  // [V, D]

  float* logits = (float*)d_out;                            // [BT, V]
  float* loss   = logits + (size_t)BT * VV;                 // scalar

  // one-time (per launch) weight convert+transpose to f16 [N][K]
  const int gDD  = (DD * DD + 255) / 256;
  const int gD4D = (DD * D4 + 255) / 256;
  for (int l = 0; l < LL; ++l) {
    _Float16* wl = wt + (size_t)l * PER;
    cvtT_kernel<<<gDD, 256, 0, stream>>>(wq  + (size_t)l * DD * DD, wl,                 DD, DD);
    cvtT_kernel<<<gDD, 256, 0, stream>>>(wk  + (size_t)l * DD * DD, wl + (size_t)DD*DD, DD, DD);
    cvtT_kernel<<<gDD, 256, 0, stream>>>(wvv + (size_t)l * DD * DD, wl + 2*(size_t)DD*DD, DD, DD);
    cvtT_kernel<<<gDD, 256, 0, stream>>>(wo  + (size_t)l * DD * DD, wl + 3*(size_t)DD*DD, DD, DD);
    cvtT_kernel<<<gD4D, 256, 0, stream>>>(w1 + (size_t)l * DD * D4, wl + 4*(size_t)DD*DD, DD, D4);
    cvtT_kernel<<<gD4D, 256, 0, stream>>>(w2 + (size_t)l * D4 * DD, wl + 4*(size_t)DD*DD + (size_t)DD*D4, D4, DD);
  }
  cvtT_kernel<<<(DD * VV + 255) / 256, 256, 0, stream>>>(lm_w, lm_wt, DD, VV);

  embed_kernel<<<(BT * DD + 255) / 256, 256, 0, stream>>>(idx, tok_emb, pos_emb, x);

  const dim3 blk(128);
  const dim3 gD(DD / 64, BT / 64);
  const dim3 g4D(D4 / 64, BT / 64);
  const dim3 gV(VV / 64, BT / 64);

  for (int l = 0; l < LL; ++l) {
    _Float16* wl  = wt + (size_t)l * PER;
    _Float16* wqt = wl;
    _Float16* wkt = wl + (size_t)DD * DD;
    _Float16* wvt = wl + 2 * (size_t)DD * DD;
    _Float16* wot = wl + 3 * (size_t)DD * DD;
    _Float16* w1t = wl + 4 * (size_t)DD * DD;                 // [4D, D]
    _Float16* w2t = w1t + (size_t)DD * D4;                    // [D, 4D]

    ln_kernel<<<BT, 256, 0, stream>>>(x, ln1_w + l * DD, ln1_b + l * DD, hb);
    gemm_kernel<false, false, false, _Float16><<<gD, blk, 0, stream>>>(
        hb, wqt, nullptr, nullptr, qb, BT, DD, DD);
    gemm_kernel<false, false, false, _Float16><<<gD, blk, 0, stream>>>(
        hb, wkt, nullptr, nullptr, kb, BT, DD, DD);
    gemm_kernel<false, false, false, _Float16><<<gD, blk, 0, stream>>>(
        hb, wvt, nullptr, nullptr, vb, BT, DD, DD);

    attn_kernel<<<dim3(TT / 64, BB * HH), blk, 0, stream>>>(qb, kb, vb, ob);

    gemm_kernel<true, true, false, float><<<gD, blk, 0, stream>>>(
        ob, wot, bo + l * DD, x, x, BT, DD, DD);

    ln_kernel<<<BT, 256, 0, stream>>>(x, ln2_w + l * DD, ln2_b + l * DD, hb);
    gemm_kernel<true, false, true, _Float16><<<g4D, blk, 0, stream>>>(
        hb, w1t, b1 + l * D4, nullptr, mb, BT, D4, DD);
    gemm_kernel<true, true, false, float><<<gD, blk, 0, stream>>>(
        mb, w2t, b2 + l * DD, x, x, BT, DD, D4);
  }

  ln_kernel<<<BT, 256, 0, stream>>>(x, lnf_w, lnf_b, hb);
  gemm_kernel<true, false, false, float><<<gV, blk, 0, stream>>>(
      hb, lm_wt, lm_b, nullptr, logits, BT, VV, DD);

  zero_kernel<<<1, 1, 0, stream>>>(loss);
  loss_kernel<<<BT, 256, 0, stream>>>(logits, targets, loss);
}
